// Mamba2Block_49495203119701
// MI455X (gfx1250) — matrix operand order repaired and checked by reading, not compiled, not run on hardware
//
#include <hip/hip_runtime.h>
#include <hip/hip_bf16.h>

// ---------------------------------------------------------------------------
// Mamba2 block forward for MI455X (gfx1250, wave32, WMMA).
// GEMMs use bf16x3 split-precision on V_WMMA_F32_16X16X32_BF16 (fp32-class
// accuracy, bf16 matrix-pipe throughput). Each wave computes a 16x32 tile so
// one A fragment feeds 6 WMMAs per K-step. Edge tiles are handled branchlessly
// (clamped row pointer + value mask) to keep EXEC full and loads vectorized.
// ---------------------------------------------------------------------------

typedef __attribute__((ext_vector_type(16))) __bf16 v16bf;
typedef __attribute__((ext_vector_type(8)))  float  v8f;

#define SEQ_LEN  1024
#define D_MODEL  768
#define D_STATE  64
#define D_CONV   4
#define N_HEADS  12
#define D_INNER  1536
#define HEAD_DIM 128
#define IN_PROJ  4620   // 2*D_INNER + 2*N_HEADS*D_STATE + N_HEADS

__device__ __forceinline__ void split_bf16(float x, __bf16 &hi, __bf16 &lo) {
    hi = (__bf16)x;
    lo = (__bf16)(x - (float)hi);
}

// C[M x N] = A[M x K] * W[N x K]^T   (all fp32 row-major, bf16x3 WMMA inside)
// One wave computes a 16x32 tile (two 16x16 WMMA tiles sharing the A frag).
template <int WPB>
__global__ __launch_bounds__(WPB * 32) void gemm_wmma_bf16x3(
    const float* __restrict__ A, const float* __restrict__ W,
    float* __restrict__ C, int M, int N, int K)
{
    const unsigned lane = threadIdx.x & 31u;
    const unsigned wave = threadIdx.x >> 5;
    const int tm  = blockIdx.x;                        // 16-row tile
    const int tn0 = (blockIdx.y * WPB + wave) * 2;     // first of two 16-col tiles
    if (tn0 * 16 >= N) return;                         // wave-uniform exit

    const int l15    = lane & 15;
    const int hiHalf = (lane & 16) ? 1 : 0;
    const int klo = hiHalf ? 8 : 0;                    // unused; A frag also uses kb
    const int kb  = hiHalf ? 16 : 0;                   // B-frag K sub-offset

    const int arow = tm * 16 + l15;                    // A row this lane loads
    const float* __restrict__ Arow = A + (size_t)arow * K;

    // Branchless N-edge handling: clamp row pointer, mask loaded values.
    const int   n0 = tn0 * 16 + l15;
    const int   n1 = n0 + 16;
    const float m0 = (n0 < N) ? 1.f : 0.f;
    const float m1 = (n1 < N) ? 1.f : 0.f;
    const float* __restrict__ W0 = W + (size_t)(n0 < N ? n0 : N - 1) * K;
    const float* __restrict__ W1 = W + (size_t)(n1 < N ? n1 : N - 1) * K;

    v8f acc0 = {0.f, 0.f, 0.f, 0.f, 0.f, 0.f, 0.f, 0.f};
    v8f acc1 = {0.f, 0.f, 0.f, 0.f, 0.f, 0.f, 0.f, 0.f};

    for (int k0 = 0; k0 < K; k0 += 32) {
        v16bf ah, al, b0h, b0l, b1h, b1l;
        // A fragment: elements 0..7 <- K = k0+kb+i ; 8..15 <- K = k0+kb+8+i
        #pragma unroll
        for (int i = 0; i < 8; ++i) {
            __bf16 h, l;
            split_bf16(Arow[k0 + kb + i], h, l);
            ah[i] = h; al[i] = l;
            split_bf16(Arow[k0 + kb + 8 + i], h, l);
            ah[8 + i] = h; al[8 + i] = l;
        }
        // B fragments: element e <- K = k0+kb+e (contiguous in W's rows)
        #pragma unroll
        for (int e = 0; e < 16; ++e) {
            __bf16 h, l;
            split_bf16(W0[k0 + kb + e] * m0, h, l);
            b0h[e] = h; b0l[e] = l;
            split_bf16(W1[k0 + kb + e] * m1, h, l);
            b1h[e] = h; b1l[e] = l;
        }
        // bf16x3: hi*hi + hi*lo + lo*hi  (lo*lo dropped; ~fp32 accuracy)
        acc0 = __builtin_amdgcn_wmma_f32_16x16x32_bf16(false, al, false, b0h,
                                                       (short)0, acc0, false, false);
        acc0 = __builtin_amdgcn_wmma_f32_16x16x32_bf16(false, ah, false, b0l,
                                                       (short)0, acc0, false, false);
        acc0 = __builtin_amdgcn_wmma_f32_16x16x32_bf16(false, ah, false, b0h,
                                                       (short)0, acc0, false, false);
        acc1 = __builtin_amdgcn_wmma_f32_16x16x32_bf16(false, al, false, b1h,
                                                       (short)0, acc1, false, false);
        acc1 = __builtin_amdgcn_wmma_f32_16x16x32_bf16(false, ah, false, b1l,
                                                       (short)0, acc1, false, false);
        acc1 = __builtin_amdgcn_wmma_f32_16x16x32_bf16(false, ah, false, b1h,
                                                       (short)0, acc1, false, false);
    }

    const int rbase = tm * 16 + (hiHalf ? 8 : 0);
    if (n0 < N) {
        #pragma unroll
        for (int r = 0; r < 8; ++r)
            C[(size_t)(rbase + r) * N + n0] = acc0[r];
    }
    if (n1 < N) {
        #pragma unroll
        for (int r = 0; r < 8; ++r)
            C[(size_t)(rbase + r) * N + n1] = acc1[r];
    }
}

// Causal conv1d(width 4) + SiLU over x_ssm slice of proj.
__global__ __launch_bounds__(256) void conv_silu_kernel(
    const float* __restrict__ proj, const float* __restrict__ conv_w,
    const float* __restrict__ conv_b, float* __restrict__ xconv)
{
    const int c = blockIdx.x * blockDim.x + threadIdx.x;   // 0..D_INNER-1
    const int t = blockIdx.y;
    if (c >= D_INNER) return;
    float acc = conv_b[c];
    #pragma unroll
    for (int i = 0; i < D_CONV; ++i) {
        int ts = t - (D_CONV - 1) + i;
        float v = (ts >= 0) ? proj[(size_t)ts * IN_PROJ + D_INNER + c] : 0.f;
        acc += v * conv_w[c * D_CONV + i];
    }
    float s = acc / (1.f + expf(-acc));        // SiLU
    xconv[(size_t)t * D_INNER + c] = s;
}

// dt = softplus(dt_raw) + dt_bias
__global__ __launch_bounds__(256) void dt_kernel(
    const float* __restrict__ proj, const float* __restrict__ dt_bias,
    float* __restrict__ dtb)
{
    const int idx = blockIdx.x * blockDim.x + threadIdx.x;
    if (idx >= SEQ_LEN * N_HEADS) return;
    const int t = idx / N_HEADS, h = idx % N_HEADS;
    float x = proj[(size_t)t * IN_PROJ + 2 * D_INNER + 2 * N_HEADS * D_STATE + h];
    float sp = (x > 20.f) ? x : log1pf(expf(x));
    dtb[idx] = sp + dt_bias[h];
}

// Sequential SSM scan. One block per head, 128 threads (= HEAD_DIM rows).
// Each lane keeps its 64-wide state row in VGPRs. Matches the reference's
// clamped-cumsum formulation: S += exp(min(-Phi,80))*dt*B*x ; y = exp(Phi)*C.S
__global__ __launch_bounds__(HEAD_DIM) void scan_kernel(
    const float* __restrict__ proj, const float* __restrict__ xconv,
    const float* __restrict__ dtb, const float* __restrict__ A_log,
    const float* __restrict__ Dp, float* __restrict__ ybuf)
{
    __shared__ float sB[D_STATE];
    __shared__ float sC[D_STATE];
    const int h = blockIdx.x;
    const int p = threadIdx.x;                 // head-dim row
    const float negA = -expf(A_log[h]);
    const float dval = Dp[h];

    float S[D_STATE];
    #pragma unroll
    for (int s = 0; s < D_STATE; ++s) S[s] = 0.f;

    float phi = 0.f;
    for (int t = 0; t < SEQ_LEN; ++t) {
        __syncthreads();                       // protect prev-iter LDS reads
        if (p < D_STATE)
            sB[p] = proj[(size_t)t * IN_PROJ + 2 * D_INNER + h * D_STATE + p];
        else
            sC[p - D_STATE] = proj[(size_t)t * IN_PROJ + 2 * D_INNER
                                   + N_HEADS * D_STATE + h * D_STATE + (p - D_STATE)];
        __syncthreads();

        const float dt = dtb[t * N_HEADS + h];
        phi += negA * dt;
        const float w = expf(fminf(-phi, 80.f));
        const float f = expf(phi);
        const float xv = xconv[(size_t)t * D_INNER + h * HEAD_DIM + p];
        const float coef = w * dt * xv;

        float acc = 0.f;
        #pragma unroll
        for (int s = 0; s < D_STATE; ++s) {
            S[s] += coef * sB[s];
            acc  += sC[s] * S[s];
        }
        ybuf[(size_t)t * D_INNER + h * HEAD_DIM + p] = f * acc + dval * xv;
    }
}

// RMSNorm over D_INNER + SiLU(z) gate, in place on ybuf.
__global__ __launch_bounds__(256) void rms_gate_kernel(
    const float* __restrict__ proj, const float* __restrict__ norm_w,
    float* __restrict__ ybuf)
{
    __shared__ float red[256];
    const int t = blockIdx.x;
    float ss = 0.f;
    for (int c = threadIdx.x; c < D_INNER; c += 256) {
        float v = ybuf[(size_t)t * D_INNER + c];
        ss += v * v;
    }
    red[threadIdx.x] = ss;
    __syncthreads();
    for (int off = 128; off > 0; off >>= 1) {
        if ((int)threadIdx.x < off) red[threadIdx.x] += red[threadIdx.x + off];
        __syncthreads();
    }
    const float rms = rsqrtf(red[0] / (float)D_INNER + 1e-5f);
    for (int c = threadIdx.x; c < D_INNER; c += 256) {
        float v = ybuf[(size_t)t * D_INNER + c];
        float z = proj[(size_t)t * IN_PROJ + c];
        float sig = 1.f / (1.f + expf(-z));
        ybuf[(size_t)t * D_INNER + c] = v * rms * norm_w[c] * (z * sig);
    }
}

extern "C" void kernel_launch(void* const* d_in, const int* in_sizes, int n_in,
                              void* d_out, int out_size, void* d_ws, size_t ws_size,
                              hipStream_t stream)
{
    const float* x       = (const float*)d_in[0];  // (1,1024,768)
    const float* W_in    = (const float*)d_in[1];  // (4620,768)
    const float* conv_w  = (const float*)d_in[2];  // (1536,4)
    const float* conv_b  = (const float*)d_in[3];  // (1536,)
    const float* A_log   = (const float*)d_in[4];  // (12,)
    const float* Dp      = (const float*)d_in[5];  // (12,)
    const float* dt_bias = (const float*)d_in[6];  // (12,)
    const float* W_out   = (const float*)d_in[7];  // (768,1536)
    const float* norm_w  = (const float*)d_in[8];  // (1536,)
    float* out = (float*)d_out;                    // (1,1024,768)

    float* ws    = (float*)d_ws;                   // ~31.6 MB total
    float* proj  = ws;                                         // 1024*4620
    float* xconv = proj  + (size_t)SEQ_LEN * IN_PROJ;          // 1024*1536
    float* dtb   = xconv + (size_t)SEQ_LEN * D_INNER;          // 1024*16 (padded)
    float* ybuf  = dtb   + (size_t)SEQ_LEN * 16;               // 1024*1536

    // 1) in-projection: proj = x @ W_in^T   (N = 4620 -> 145 pair-tiles)
    {
        const int pairTiles = (IN_PROJ + 31) / 32;             // 145
        dim3 grid(SEQ_LEN / 16, (pairTiles + 3) / 4);
        gemm_wmma_bf16x3<4><<<grid, 128, 0, stream>>>(
            x, W_in, proj, SEQ_LEN, IN_PROJ, D_MODEL);
    }
    // 2) conv + SiLU, and dt = softplus + bias
    {
        dim3 grid(D_INNER / 256, SEQ_LEN);
        conv_silu_kernel<<<grid, 256, 0, stream>>>(proj, conv_w, conv_b, xconv);
        dt_kernel<<<(SEQ_LEN * N_HEADS + 255) / 256, 256, 0, stream>>>(
            proj, dt_bias, dtb);
    }
    // 3) sequential SSM scan (one block per head)
    scan_kernel<<<N_HEADS, HEAD_DIM, 0, stream>>>(
        proj, xconv, dtb, A_log, Dp, ybuf);
    // 4) RMSNorm + SiLU(z) gate, in place
    rms_gate_kernel<<<SEQ_LEN, 256, 0, stream>>>(proj, norm_w, ybuf);
    // 5) out-projection: out = ybuf @ W_out^T  (N = 768 -> 24 pair-tiles)
    {
        dim3 grid(SEQ_LEN / 16, 24 / 4);
        gemm_wmma_bf16x3<4><<<grid, 128, 0, stream>>>(
            ybuf, W_out, out, SEQ_LEN, D_MODEL, D_INNER);
    }
}
